// Attention_24352464569797
// MI455X (gfx1250) — matrix operand order repaired and checked by reading, not compiled, not run on hardware
//
#include <hip/hip_runtime.h>
#include <hip/hip_bf16.h>
#include <stdint.h>

// ---------------------------------------------------------------------------
// MHA for MI455X (gfx1250): f16 WMMA everywhere, async global->LDS (ASYNCcnt)
// double-buffered staging for GEMMs and flash-attention K/V tiles.
// B=2, S=4096, D=768, H=12, HD=64.
// ---------------------------------------------------------------------------

typedef __attribute__((ext_vector_type(16))) _Float16 v16h;
typedef __attribute__((ext_vector_type(8)))  _Float16 v8h;
typedef __attribute__((ext_vector_type(8)))  float    v8f;

#define B_   2
#define S_   4096
#define D_   768
#define H_   12
#define HD_  64

__device__ __forceinline__ v8f wmma32(v16h a, v16h b, v8f c) {
  return __builtin_amdgcn_wmma_f32_16x16x32_f16(
      /*neg_a=*/false, a, /*neg_b=*/false, b,
      /*c_mod=*/(short)0, c, /*reuse_a=*/false, /*reuse_b=*/false);
}

// A-fragment (16x32, M x K), row-major source (leading dim in halves).
// lane l (m=l&15, h=l>>4): VGPR0-3 hold K=8h..8h+7, VGPR4-7 hold K=16+8h..23+8h.
__device__ __forceinline__ v16h load_frag_a(const _Float16* base, int ld) {
  const int l = threadIdx.x & 31;
  const int m = l & 15, h = l >> 4;
  const _Float16* p = base + (long)m * ld;
  v8h lo = *(const v8h*)(p + 8 * h);
  v8h hi = *(const v8h*)(p + 16 + 8 * h);
  v16h r;
#pragma unroll
  for (int i = 0; i < 8; ++i) { r[i] = lo[i]; r[8 + i] = hi[i]; }
  return r;
}

// B-fragment (32x16, K x N) fed from Bt = B^T stored [N][K] row-major.
// lane l (n=l&15, h=l>>4): VGPR0-3 hold K=8h..8h+7, VGPR4-7 hold K=16+8h..23+8h.
__device__ __forceinline__ v16h load_frag_b(const _Float16* baseT, int ldk) {
  const int l = threadIdx.x & 31;
  const int n = l & 15, h = l >> 4;
  const _Float16* p = baseT + (long)n * ldk;
  v8h lo = *(const v8h*)(p + 8 * h);
  v8h hi = *(const v8h*)(p + 16 + 8 * h);
  v16h r;
#pragma unroll
  for (int i = 0; i < 8; ++i) { r[i] = lo[i]; r[8 + i] = hi[i]; }
  return r;
}

// CDNA5 async global -> LDS copy, 16 bytes per lane (ASYNCcnt-tracked).
__device__ __forceinline__ void async_copy_b128(void* lds_ptr, const void* gptr) {
  const uint32_t lds_off = (uint32_t)(uintptr_t)lds_ptr;
  const unsigned long long ga = (unsigned long long)(uintptr_t)gptr;
  asm volatile("global_load_async_to_lds_b128 %0, %1, off"
               :: "v"(lds_off), "v"(ga) : "memory");
}
__device__ __forceinline__ void wait_async4(void) {
  asm volatile("s_wait_asynccnt 0x4" ::: "memory");
}
__device__ __forceinline__ void wait_async0(void) {
  asm volatile("s_wait_asynccnt 0x0" ::: "memory");
}

// ---------------------------------------------------------------------------
// Converters
// ---------------------------------------------------------------------------
__global__ void cvt_f32_to_f16(const float* __restrict__ in,
                               _Float16* __restrict__ out, int n) {
  for (int i = blockIdx.x * 256 + threadIdx.x; i < n; i += gridDim.x * 256)
    out[i] = (_Float16)in[i];
}

__global__ void transpose_w_f16(const float* __restrict__ w,
                                _Float16* __restrict__ wt, int K, int N) {
  for (int i = blockIdx.x * 256 + threadIdx.x; i < K * N; i += gridDim.x * 256) {
    int n = i / K, k = i % K;
    wt[i] = (_Float16)w[(long)k * N + n];
  }
}

// ---------------------------------------------------------------------------
// GEMM core: C[128x128 block] = A[8192x768] @ Bt^T, operands staged in LDS
// via async global->LDS, double-buffered, k-loop unrolled x2.
// 256 threads = 8 waves; wave tile 32(m) x 64(n) = 8 WMMAs per k-step.
// The epilogue differs between projection (f16, head-split) and output (f32),
// so the core is a macro-ish inline with accumulators returned in-place.
// ---------------------------------------------------------------------------
struct GemmAcc { v8f acc[2][4]; };

__device__ __forceinline__ void gemm_core(
    const _Float16* __restrict__ Ag,   // A block base: row (blockm*128), col 0
    const _Float16* __restrict__ Bg,   // Bt block base: row (blockn*128), col 0
    _Float16 (*Al)[128][32], _Float16 (*Bl)[128][32],
    GemmAcc& g) {
  const int tid  = threadIdx.x;
  const int wave = tid >> 5;
  const int wm = (wave & 3) * 32;      // wave m offset in block tile
  const int wn = (wave >> 2) * 64;     // wave n offset in block tile

#pragma unroll
  for (int mi = 0; mi < 2; ++mi)
#pragma unroll
    for (int t = 0; t < 4; ++t)
#pragma unroll
      for (int i = 0; i < 8; ++i) g.acc[mi][t][i] = 0.f;

  // cooperative async fill of one 128x32 A tile + 128x32 B tile
  auto issue_tile = [&](int buf, int kk) {
#pragma unroll
    for (int it = 0; it < 2; ++it) {
      const int c = tid + it * 256;          // 512 x 16B chunks each matrix
      const int r = c >> 2, co = (c & 3) * 8;
      async_copy_b128(&Al[buf][r][co], Ag + (long)r * D_ + kk + co);
      async_copy_b128(&Bl[buf][r][co], Bg + (long)r * D_ + kk + co);
    }
  };

  auto mma_step = [&](int buf) {
    v16h a0 = load_frag_a(&Al[buf][wm][0], 32);
    v16h a1 = load_frag_a(&Al[buf][wm + 16][0], 32);
    v16h bfr[4];
#pragma unroll
    for (int t = 0; t < 4; ++t)
      bfr[t] = load_frag_b(&Bl[buf][wn + t * 16][0], 32);
#pragma unroll
    for (int t = 0; t < 4; ++t) {
      g.acc[0][t] = wmma32(a0, bfr[t], g.acc[0][t]);
      g.acc[1][t] = wmma32(a1, bfr[t], g.acc[1][t]);
    }
  };

  issue_tile(0, 0);
  for (int k = 0; k < D_; k += 64) {
    issue_tile(1, k + 32);               // always in range (k <= 704)
    wait_async4();                       // retire buf0's 4 issues (in-order)
    __syncthreads();
    mma_step(0);
    __syncthreads();                     // done with buf0 before refill
    if (k + 64 < D_) {
      issue_tile(0, k + 64);
      wait_async4();
    } else {
      wait_async0();
    }
    __syncthreads();
    mma_step(1);
    __syncthreads();
  }
}

// ---------------------------------------------------------------------------
// Projection GEMM: Y = Xh @ W + b -> f16, head-split layout.
// mode 0: out[b][h][s][hd]   (Q,K)    mode 1: out[b][h][hd][s]   (V)
// Grid (64, 6), 256 threads.
// ---------------------------------------------------------------------------
__global__ __launch_bounds__(256) void proj_gemm(
    const _Float16* __restrict__ A, const _Float16* __restrict__ Bt,
    const float* __restrict__ bias, _Float16* __restrict__ out,
    int mode, float scale) {
  __shared__ _Float16 Al[2][128][32];   // 8 KB each buffer
  __shared__ _Float16 Bl[2][128][32];

  const int mblk = blockIdx.x * 128;
  const int nblk = blockIdx.y * 128;
  GemmAcc g;
  gemm_core(A + (long)mblk * D_, Bt + (long)nblk * D_, Al, Bl, g);

  const int wave = threadIdx.x >> 5;
  const int mbase = mblk + (wave & 3) * 32;
  const int nbase = nblk + (wave >> 2) * 64;
  const int l = threadIdx.x & 31;
  const int n0 = l & 15, half = l >> 4;
#pragma unroll
  for (int mi = 0; mi < 2; ++mi) {
#pragma unroll
    for (int t = 0; t < 4; ++t) {
      const int col = nbase + t * 16 + n0;
      const float bs = bias[col];
      const int h  = col >> 6;
      const int hd = col & 63;
#pragma unroll
      for (int r = 0; r < 8; ++r) {
        const int m = mbase + mi * 16 + r + 8 * half;  // row = b*S + s
        const int bb = m >> 12, s = m & (S_ - 1);
        const float v = (g.acc[mi][t][r] + bs) * scale;
        long off;
        if (mode == 0) off = (((long)(bb * H_ + h) * S_ + s) * HD_ + hd);
        else           off = (((long)(bb * H_ + h) * HD_ + hd) * S_ + s);
        out[off] = (_Float16)v;
      }
    }
  }
}

// ---------------------------------------------------------------------------
// Output projection: out (f32) = ctx @ Wo + bo.
// ---------------------------------------------------------------------------
__global__ __launch_bounds__(256) void out_gemm(
    const _Float16* __restrict__ A, const _Float16* __restrict__ Bt,
    const float* __restrict__ bias, float* __restrict__ out) {
  __shared__ _Float16 Al[2][128][32];
  __shared__ _Float16 Bl[2][128][32];

  const int mblk = blockIdx.x * 128;
  const int nblk = blockIdx.y * 128;
  GemmAcc g;
  gemm_core(A + (long)mblk * D_, Bt + (long)nblk * D_, Al, Bl, g);

  const int wave = threadIdx.x >> 5;
  const int mbase = mblk + (wave & 3) * 32;
  const int nbase = nblk + (wave >> 2) * 64;
  const int l = threadIdx.x & 31;
  const int n0 = l & 15, half = l >> 4;
#pragma unroll
  for (int mi = 0; mi < 2; ++mi) {
#pragma unroll
    for (int t = 0; t < 4; ++t) {
      const int col = nbase + t * 16 + n0;
      const float bs = bias[col];
#pragma unroll
      for (int r = 0; r < 8; ++r) {
        const int m = mbase + mi * 16 + r + 8 * half;
        out[(long)m * D_ + col] = g.acc[mi][t][r] + bs;
      }
    }
  }
}

// ---------------------------------------------------------------------------
// Flash attention.
// Block = 4 waves = 64 query rows of one (b,h); each wave one 16-row tile.
// K/V tiles (32 keys) staged via async global->LDS, double-buffered;
// j-loop unrolled x2 so the LDS buffer index is a compile-time constant.
// Q pre-scaled by 1/sqrt(HD). K in [B,H,S,HD], V transposed [B,H,HD,S].
// ---------------------------------------------------------------------------
__global__ __launch_bounds__(128) void flash_attn(
    const _Float16* __restrict__ Qh, const _Float16* __restrict__ Kh,
    const _Float16* __restrict__ Vt, const float* __restrict__ mask,
    _Float16* __restrict__ ctx) {
  __shared__ _Float16 Kt [2][32][64];   // [buf][key][hd]      4 KB each
  __shared__ _Float16 Vtl[2][64][32];   // [buf][hd][key]      4 KB each
  __shared__ _Float16 Pb [4][16][32];   // per-wave P staging  1 KB each

  const int wid = threadIdx.x >> 5;
  const int tid = threadIdx.x;
  const int b  = blockIdx.x / (H_ * 64);
  const int h  = (blockIdx.x / 64) % H_;
  const int qg = blockIdx.x & 63;          // 64 query groups of 64 rows
  const int q0 = (qg * 4 + wid) * 16;

  const long head = (long)(b * H_ + h) * S_ * HD_;
  const _Float16* Qp = Qh + head + (long)q0 * HD_;
  const _Float16* Kp = Kh + head;
  const _Float16* Vp = Vt + head;          // [HD][S]
  const float*    mk = mask + (long)b * S_;

  // cooperative async fill of one 32-key K/V tile (4 b128 issues per thread)
  auto issue_tile = [&](int buf, int j) {
#pragma unroll
    for (int it = 0; it < 2; ++it) {
      const int c = tid + it * 128;            // 256 x 16B chunks each
      const int kr = c >> 3, kc = (c & 7) * 8;
      async_copy_b128(&Kt[buf][kr][kc], Kp + (long)(j + kr) * HD_ + kc);
      const int vr = c >> 2, vc = (c & 3) * 8;
      async_copy_b128(&Vtl[buf][vr][vc], Vp + (long)vr * S_ + j + vc);
    }
  };

  issue_tile(0, 0);  // prologue: start filling buffer 0 immediately

  const int l = threadIdx.x & 31;
  const int lane_n = l & 15, half = l >> 4;

  v16h qa0 = load_frag_a(Qp, HD_);
  v16h qa1 = load_frag_a(Qp + 32, HD_);

  v8f O[4];
#pragma unroll
  for (int t = 0; t < 4; ++t)
#pragma unroll
    for (int i = 0; i < 8; ++i) O[t][i] = 0.f;
  float M[8], L[8];
#pragma unroll
  for (int r = 0; r < 8; ++r) { M[r] = -1e30f; L[r] = 0.f; }

  // one 32-key step out of buffer `cur` (cur is a literal at call sites)
  auto compute_tile = [&](int cur, int j) {
    v8f s0, s1;
#pragma unroll
    for (int i = 0; i < 8; ++i) { s0[i] = 0.f; s1[i] = 0.f; }
    {
      v16h b00 = load_frag_b(&Kt[cur][ 0][ 0], HD_);
      v16h b01 = load_frag_b(&Kt[cur][ 0][32], HD_);
      v16h b10 = load_frag_b(&Kt[cur][16][ 0], HD_);
      v16h b11 = load_frag_b(&Kt[cur][16][32], HD_);
      s0 = wmma32(qa0, b00, s0);
      s0 = wmma32(qa1, b01, s0);
      s1 = wmma32(qa0, b10, s1);
      s1 = wmma32(qa1, b11, s1);
    }
    const float madd0 = -1e6f * (1.f - mk[j + lane_n]);
    const float madd1 = -1e6f * (1.f - mk[j + 16 + lane_n]);

    float rscale[8];
#pragma unroll
    for (int r = 0; r < 8; ++r) {
      float a0 = s0[r] + madd0;
      float a1 = s1[r] + madd1;
      float mx = fmaxf(a0, a1);
      mx = fmaxf(mx, __shfl_xor(mx, 1, 32));
      mx = fmaxf(mx, __shfl_xor(mx, 2, 32));
      mx = fmaxf(mx, __shfl_xor(mx, 4, 32));
      mx = fmaxf(mx, __shfl_xor(mx, 8, 32));
      const float mnew = fmaxf(M[r], mx);
      const float sc = __expf(M[r] - mnew);
      const float p0 = __expf(a0 - mnew);
      const float p1 = __expf(a1 - mnew);
      float rs = p0 + p1;
      rs += __shfl_xor(rs, 1, 32);
      rs += __shfl_xor(rs, 2, 32);
      rs += __shfl_xor(rs, 4, 32);
      rs += __shfl_xor(rs, 8, 32);
      L[r] = L[r] * sc + rs;
      M[r] = mnew;
      rscale[r] = sc;
      Pb[wid][r + 8 * half][lane_n]      = (_Float16)p0;
      Pb[wid][r + 8 * half][16 + lane_n] = (_Float16)p1;
    }
    asm volatile("s_wait_dscnt 0x0" ::: "memory");  // P visible within wave

#pragma unroll
    for (int t = 0; t < 4; ++t)
#pragma unroll
      for (int r = 0; r < 8; ++r) O[t][r] *= rscale[r];

    v16h pa = load_frag_a(&Pb[wid][0][0], 32);
    v16h vb[4];
#pragma unroll
    for (int t = 0; t < 4; ++t)
      vb[t] = load_frag_b(&Vtl[cur][t * 16][0], 32);
#pragma unroll
    for (int t = 0; t < 4; ++t)
      O[t] = wmma32(pa, vb[t], O[t]);
  };

  for (int j = 0; j < S_; j += 64) {
    // ---- body A: buffer 0 holds keys [j, j+32) ----
    issue_tile(1, j + 32);       // always in range (j <= S-64)
    wait_async4();               // retire buffer 0's 4 issues (in-order)
    __syncthreads();
    compute_tile(0, j);
    __syncthreads();             // done reading buf0 before its refill

    // ---- body B: buffer 1 holds keys [j+32, j+64) ----
    if (j + 64 < S_) {
      issue_tile(0, j + 64);
      wait_async4();
    } else {
      wait_async0();
    }
    __syncthreads();
    compute_tile(1, j + 32);
    __syncthreads();
  }

  // epilogue: normalize, write ctx[b][s][h*64+hd] (f16)
#pragma unroll
  for (int t = 0; t < 4; ++t) {
#pragma unroll
    for (int r = 0; r < 8; ++r) {
      const float v = O[t][r] / L[r];
      const int s = q0 + r + 8 * half;
      const long off = ((long)b * S_ + s) * D_ + h * HD_ + t * 16 + lane_n;
      ctx[off] = (_Float16)v;
    }
  }
}

// ---------------------------------------------------------------------------
extern "C" void kernel_launch(void* const* d_in, const int* in_sizes, int n_in,
                              void* d_out, int out_size, void* d_ws,
                              size_t ws_size, hipStream_t stream) {
  const float* X    = (const float*)d_in[0];
  const float* mask = (const float*)d_in[1];
  const float* Wq   = (const float*)d_in[2];
  const float* bq   = (const float*)d_in[3];
  const float* Wk   = (const float*)d_in[4];
  const float* bk   = (const float*)d_in[5];
  const float* Wv   = (const float*)d_in[6];
  const float* bv   = (const float*)d_in[7];
  const float* Wo   = (const float*)d_in[8];
  const float* bo   = (const float*)d_in[9];
  float* out = (float*)d_out;

  const size_t XN = (size_t)B_ * S_ * D_;
  const size_t WN = (size_t)D_ * D_;
  const size_t QN = (size_t)B_ * H_ * S_ * HD_;

  char* ws = (char*)d_ws;
  _Float16* Xh  = (_Float16*)ws; ws += XN * 2;
  _Float16* Wqt = (_Float16*)ws; ws += WN * 2;
  _Float16* Wkt = (_Float16*)ws; ws += WN * 2;
  _Float16* Wvt = (_Float16*)ws; ws += WN * 2;
  _Float16* Wot = (_Float16*)ws; ws += WN * 2;
  _Float16* Qh  = (_Float16*)ws; ws += QN * 2;
  _Float16* Kh  = (_Float16*)ws; ws += QN * 2;
  _Float16* Vt  = (_Float16*)ws; ws += QN * 2;
  _Float16* Ctx = (_Float16*)ws; ws += QN * 2;
  (void)ws_size; (void)in_sizes; (void)n_in; (void)out_size;

  cvt_f32_to_f16<<<(int)((XN + 255) / 256), 256, 0, stream>>>(X, Xh, (int)XN);
  transpose_w_f16<<<(int)((WN + 255) / 256), 256, 0, stream>>>(Wq, Wqt, D_, D_);
  transpose_w_f16<<<(int)((WN + 255) / 256), 256, 0, stream>>>(Wk, Wkt, D_, D_);
  transpose_w_f16<<<(int)((WN + 255) / 256), 256, 0, stream>>>(Wv, Wvt, D_, D_);
  transpose_w_f16<<<(int)((WN + 255) / 256), 256, 0, stream>>>(Wo, Wot, D_, D_);

  dim3 ggrid(64, 6);  // 8192/128 x 768/128
  const float qscale = 0.125f;  // 1/sqrt(64) folded into Q
  proj_gemm<<<ggrid, 256, 0, stream>>>(Xh, Wqt, bq, Qh, 0, qscale);
  proj_gemm<<<ggrid, 256, 0, stream>>>(Xh, Wkt, bk, Kh, 0, 1.0f);
  proj_gemm<<<ggrid, 256, 0, stream>>>(Xh, Wvt, bv, Vt, 1, 1.0f);

  // 2 * 12 * 64 query-groups (64 rows each), 4 waves per block
  flash_attn<<<B_ * H_ * 64, 128, 0, stream>>>(Qh, Kh, Vt, mask, Ctx);

  out_gemm<<<ggrid, 256, 0, stream>>>(Ctx, Wot, bo, out);
}
